// TopKTopPSampler_28449863369160
// MI455X (gfx1250) — matrix operand order repaired and checked by reading, not compiled, not run on hardware
//
#include <hip/hip_runtime.h>
#include <stdint.h>

// ---------------------------------------------------------------------------
// Top-k + top-p + Gumbel-max sampling, one workgroup per row.
//   B = 128 rows, V = 128000 logits/row.
// Pass 1: 65536-bin radix histogram of high-16 monotonic float bits (256KB LDS,
//         exploiting CDNA5's 320KB WGP LDS), streamed via TDM double buffering.
// Pass 2: collect all candidates with bucket >= k-th bucket (<= ~2K values).
// Then in-LDS: bitonic sort, f16 softmax weights, WMMA-based prefix scan for
// the top-p cumulative cutoff, gather q at surviving indices, argmax.
// ---------------------------------------------------------------------------

#define B_ROWS     128
#define V_COLS     128000
#define NTHREADS   256
#define TILE_ELEMS 4000
#define TILE_BYTES (TILE_ELEMS * 4)
#define NTILES     32          // 32 * 4000 = 128000
#define CAP        2048
#define KMAX       1024
#define NBINS      65536

// LDS layout (bytes). hist region is reused for candidate/scan arrays.
#define OFF_CANDVAL 0          // float[2048]
#define OFF_CANDIDX 8192       // uint [2048]
#define OFF_WH      16384      // f16  [1024]
#define OFF_INCL    18432      // float[1024]
#define OFF_SEGOFF  22528      // float[32]
#define OFF_SCAL    22656      // struct Scal
#define OFF_HIST    0          // uint [65536] = 256KB
#define OFF_CHUNK   262144     // uint [256] (dead before staging reuse matters)
#define OFF_STAGE   262144     // 2 x 16000B staging buffers
#define SMEM_BYTES  (OFF_STAGE + 2 * TILE_BYTES)   // 294144 <= 320KB

typedef _Float16 f16;
typedef __attribute__((ext_vector_type(16))) _Float16     v16h;
typedef __attribute__((ext_vector_type(8)))  float        v8f;
typedef __attribute__((ext_vector_type(4)))  unsigned int u32x4;
typedef __attribute__((ext_vector_type(8)))  int          i32x8;
typedef __attribute__((ext_vector_type(4)))  int          i32x4;

#if defined(__HIP_DEVICE_COMPILE__)
  #if __has_builtin(__builtin_amdgcn_tensor_load_to_lds) && \
      __has_builtin(__builtin_amdgcn_s_wait_tensorcnt)
    #define USE_TDM 1
  #else
    #define USE_TDM 0
  #endif
  #if __has_builtin(__builtin_amdgcn_wmma_f32_16x16x32_f16)
    #define USE_WMMA 1
  #else
    #define USE_WMMA 0
  #endif
#else
  #define USE_TDM 0
  #define USE_WMMA 0
#endif

#if __has_include(<hip/amd_detail/amd_gfx1250_TDM.h>)
  #define TDM_6ARG 1   // amdgpu-toolchain (clang-23) 6-arg builtin form
#endif

struct Scal {
  unsigned Bk;                 // selected high-16 bucket threshold
  unsigned cnt;                // candidate counter
  float    Z;                  // total softmax mass of top-k set (f16-consistent)
  unsigned pad;
  unsigned long long best;     // packed argmax key
};

// Order-preserving map: float bits -> unsigned (ascending).
__device__ inline unsigned mono_u32(unsigned b) {
  return (b & 0x80000000u) ? ~b : (b | 0x80000000u);
}

#if USE_TDM
// Issue a 1-D TDM copy of nElems f32 from global to LDS byte offset ldsOff.
__device__ inline void tdm_issue_1d(const float* gsrc, unsigned ldsOff, int nElems) {
  uint64_t ga = (uint64_t)(uintptr_t)gsrc;
  unsigned n  = (unsigned)nElems;
  u32x4 g0;
  g0[0] = 1u;                                              // count=1, no gather
  g0[1] = ldsOff;                                          // lds_addr (bytes)
  g0[2] = (unsigned)(ga & 0xFFFFFFFFull);                  // global_addr[31:0]
  g0[3] = (unsigned)((ga >> 32) & 0x01FFFFFFull)           // global_addr[56:32]
          | (2u << 30);                                    // type=2 ("image")
  i32x8 g1;
  g1[0] = (int)(2u << 16);                                 // data_size=4B, mask=0
  g1[1] = (int)((n & 0xFFFFu) << 16);                      // tensor_dim0[15:0]
  g1[2] = (int)(((n >> 16) & 0xFFFFu) | (1u << 16));       // dim0[31:16], dim1=1
  g1[3] = (int)((n & 0xFFFFu) << 16);                      // tile_dim0 = n
  g1[4] = 0;                                               // tile_dim1/2 unused
  g1[5] = (int)n;                                          // tensor_dim0_stride lo
  g1[6] = 0;
  g1[7] = 0;
  i32x4 z4 = {0, 0, 0, 0};
#if defined(TDM_6ARG)
  i32x8 z8 = {0, 0, 0, 0, 0, 0, 0, 0};
  __builtin_amdgcn_tensor_load_to_lds(g0, g1, z4, z4, z8, 0);
#else
  __builtin_amdgcn_tensor_load_to_lds(g0, g1, z4, z4, 0);
#endif
}
#endif

// Stream one row of logits through the block, calling f(value, index).
// TDM path: wave 0 DMAs tile t+1 into LDS while all waves consume tile t.
template <typename F>
__device__ inline void stream_row(const float* __restrict__ rowPtr,
                                  unsigned char* smem, F f) {
#if USE_TDM
  const bool w0 = (threadIdx.x < 32);   // wave-uniform branch
  if (w0) tdm_issue_1d(rowPtr, OFF_STAGE, TILE_ELEMS);
  for (int t = 0; t < NTILES; ++t) {
    if (w0) {
      if (t + 1 < NTILES) {
        tdm_issue_1d(rowPtr + (size_t)(t + 1) * TILE_ELEMS,
                     OFF_STAGE + (unsigned)((t + 1) & 1) * TILE_BYTES,
                     TILE_ELEMS);
        __builtin_amdgcn_s_wait_tensorcnt(1);   // tile t landed
      } else {
        __builtin_amdgcn_s_wait_tensorcnt(0);
      }
    }
    __syncthreads();
    const float* buf = (const float*)(smem + OFF_STAGE + (size_t)(t & 1) * TILE_BYTES);
    const int base = t * TILE_ELEMS;
    for (int j = threadIdx.x; j < TILE_ELEMS; j += NTHREADS)
      f(buf[j], base + j);
    __syncthreads();
  }
#else
  for (int i = threadIdx.x; i < V_COLS; i += NTHREADS)
    f(rowPtr[i], i);
  __syncthreads();
#endif
}

__global__ __launch_bounds__(NTHREADS)
void topk_topp_sample_kernel(const float* __restrict__ logits,
                             const int*   __restrict__ kArr,
                             const float* __restrict__ pArr,
                             const float* __restrict__ qArr,
                             float*       __restrict__ out) {
  extern __shared__ unsigned char smem[];
  const int row = blockIdx.x;
  const int tid = threadIdx.x;
  const float* rowLogits = logits + (size_t)row * V_COLS;

  unsigned* hist    = (unsigned*)(smem + OFF_HIST);
  unsigned* chunk   = (unsigned*)(smem + OFF_CHUNK);
  float*    candVal = (float*)(smem + OFF_CANDVAL);
  unsigned* candIdx = (unsigned*)(smem + OFF_CANDIDX);
  f16*      wh      = (f16*)(smem + OFF_WH);
  float*    incl    = (float*)(smem + OFF_INCL);
  float*    segOff  = (float*)(smem + OFF_SEGOFF);
  Scal*     sc      = (Scal*)(smem + OFF_SCAL);

  int ktgt = kArr[row];
  if (ktgt < 1)        ktgt = 1;
  if (ktgt > KMAX - 1) ktgt = KMAX - 1;

  // ---- phase 0: clear histogram ------------------------------------------
  for (int i = tid; i < NBINS; i += NTHREADS) hist[i] = 0u;
  __syncthreads();

  // ---- phase 1: 16-bit-bucket histogram of the row -----------------------
  stream_row(rowLogits, smem, [&](float v, int) {
    unsigned u = mono_u32(__float_as_uint(v));
    atomicAdd(&hist[u >> 16], 1u);
  });

  // ---- phase 1.5: find bucket containing the k-th largest ----------------
  {
    unsigned s = 0;
    const int c = tid;                      // 256 chunks of 256 bins
    for (int b = 0; b < 256; ++b) s += hist[c * 256 + b];
    chunk[c] = s;
  }
  __syncthreads();
  if (tid == 0) {
    long long need = ktgt;
    long long cum  = 0;
    int C = 255;
    for (; C > 0; --C) {
      if (cum + (long long)chunk[C] >= need) break;
      cum += chunk[C];
    }
    unsigned bk = (unsigned)(C * 256);
    for (int b = 255; b >= 0; --b) {
      unsigned h = hist[C * 256 + b];
      if (cum + (long long)h >= need) { bk = (unsigned)(C * 256 + b); break; }
      cum += h;
    }
    sc->Bk   = bk;
    sc->cnt  = 0u;
    sc->best = 0ull;
  }
  __syncthreads();

  // ---- phase 2: collect candidates with bucket >= Bk ---------------------
  const unsigned bk = sc->Bk;
  __syncthreads();
  stream_row(rowLogits, smem, [&](float v, int gi) {
    unsigned u = mono_u32(__float_as_uint(v));
    if ((u >> 16) >= bk) {
      unsigned pos = atomicAdd(&sc->cnt, 1u);
      if (pos < CAP) { candVal[pos] = v; candIdx[pos] = (unsigned)gi; }
    }
  });

  int n = (int)sc->cnt;
  if (n > CAP) n = CAP;
  const float NEG_INF = -__builtin_inff();
  for (int i = n + tid; i < CAP; i += NTHREADS) {
    candVal[i] = NEG_INF;
    candIdx[i] = 0u;
  }
  __syncthreads();

  // ---- phase 3: bitonic sort (descending) of CAP candidates --------------
  for (int size = 2; size <= CAP; size <<= 1) {
    for (int stride = size >> 1; stride > 0; stride >>= 1) {
      for (int i = tid; i < CAP; i += NTHREADS) {
        int j2 = i ^ stride;
        if (j2 > i) {
          bool desc = ((i & size) == 0);
          float vi = candVal[i], vj = candVal[j2];
          if ((vi < vj) == desc) {
            candVal[i] = vj; candVal[j2] = vi;
            unsigned ti = candIdx[i]; candIdx[i] = candIdx[j2]; candIdx[j2] = ti;
          }
        }
      }
      __syncthreads();
    }
  }

  // ---- phase 4: f16 softmax weights of top-k set -------------------------
  int kk = ktgt < n ? ktgt : n;
  if (kk < 1) kk = 1;
  const float m = candVal[0];
  for (int i = tid; i < KMAX; i += NTHREADS)
    wh[i] = (i < kk) ? (f16)expf(candVal[i] - m) : (f16)0.f;
  __syncthreads();

  // ---- phase 5: prefix scan of wh via WMMA -------------------------------
  // A (16x32 f16): row i = 32-element segment of wh. B (32x16 f16): ones with
  // B[k][j] = (k <= 2j+1).  D[i][j] (f32) = inclusive prefix of segment i at
  // odd position 2j+1.  Waves 0 and 1 each cover 512 elements.
  {
    const int wv = tid >> 5;
#if USE_WMMA
    if (wv < 2) {
      const int lane = tid & 31;
      const int Mi   = lane & 15;
      const int jj   = lane & 15;
      const int koff = (lane < 16) ? 0 : 16;
      const int base = wv * 512;
      v16h a, b;
      #pragma unroll
      for (int v = 0; v < 8; ++v) {
        int Kb = 2 * v + koff;
        a[2 * v]     = wh[base + Mi * 32 + Kb];
        a[2 * v + 1] = wh[base + Mi * 32 + Kb + 1];
        // B: lanes 0..15 hold rows K=2v,2v+1 at col jj; lanes 16..31 rows K=16+2v.
        f16 bv;
        if (lane < 16) bv = (v <= jj)     ? (f16)1.f : (f16)0.f;
        else           bv = (v <= jj - 8) ? (f16)1.f : (f16)0.f;
        b[2 * v]     = bv;
        b[2 * v + 1] = bv;
      }
      v8f c = 0.0f;
      v8f d = __builtin_amdgcn_wmma_f32_16x16x32_f16(
          /*neg_a=*/false, a, /*neg_b=*/false, b,
          /*c_mod=*/(short)0, c, /*reuse_a=*/false, /*reuse_b=*/false);
      #pragma unroll
      for (int r = 0; r < 8; ++r) {
        int seg = wv * 16 + ((lane < 16) ? r : r + 8);
        incl[seg * 32 + 2 * jj + 1] = d[r];
      }
    }
#else
    if (wv == 0) {   // scalar fallback: per-segment serial scan by 32 lanes
      const int lane = tid & 31;
      float run = 0.f;
      for (int j = 0; j < 32; ++j) {
        run += (float)wh[lane * 32 + j];
        incl[lane * 32 + j] = run;
      }
    }
#endif
  }
  __syncthreads();

  // ---- phase 6: even-position fixup + cross-segment offsets --------------
  for (int i = tid; i < 512; i += NTHREADS) {
    int pos = 2 * i;
    incl[pos] = incl[pos + 1] - (float)wh[pos + 1];
  }
  __syncthreads();
  if (tid == 0) {
    float run = 0.f;
    for (int s = 0; s < 32; ++s) {
      segOff[s] = run;
      run += incl[s * 32 + 31];
    }
    sc->Z = run;
  }
  __syncthreads();
  for (int i = tid; i < KMAX; i += NTHREADS)
    incl[i] += segOff[i >> 5];
  __syncthreads();

  // ---- phase 7: top-p keep + Gumbel-max argmax ---------------------------
  const float pZ = pArr[row] * sc->Z;
  const size_t rowBase = (size_t)row * V_COLS;
  for (int i = tid; i < kk; i += NTHREADS) {
    float excl = incl[i] - (float)wh[i];   // mass strictly above this element
    if (i == 0) excl = 0.f;                // always keep the max logit
    if (excl < pZ) {
      unsigned gi = candIdx[i];
      float score = candVal[i] - logf(qArr[rowBase + gi]);
      unsigned mk = mono_u32(__float_as_uint(score));
      unsigned long long key =
          ((unsigned long long)mk << 32) | (unsigned long long)(0xFFFFFFFFu - gi);
      atomicMax((unsigned long long*)&sc->best, key);
    }
  }
  __syncthreads();
  if (tid == 0) {
    unsigned gi = 0xFFFFFFFFu - (unsigned)(sc->best & 0xFFFFFFFFull);
    if (sc->best == 0ull) gi = candIdx[0];  // degenerate guard
    out[row] = (float)gi;
  }
}

extern "C" void kernel_launch(void* const* d_in, const int* in_sizes, int n_in,
                              void* d_out, int out_size, void* d_ws, size_t ws_size,
                              hipStream_t stream) {
  (void)in_sizes; (void)n_in; (void)out_size; (void)d_ws; (void)ws_size;
  const float* logits = (const float*)d_in[0];
  const int*   k      = (const int*)d_in[1];
  const float* p      = (const float*)d_in[2];
  const float* q      = (const float*)d_in[3];
  float*       out    = (float*)d_out;

  (void)hipFuncSetAttribute((const void*)topk_topp_sample_kernel,
                            hipFuncAttributeMaxDynamicSharedMemorySize,
                            SMEM_BYTES);
  topk_topp_sample_kernel<<<B_ROWS, NTHREADS, SMEM_BYTES, stream>>>(
      logits, k, p, q, out);
}
